// PVMLayer_72387378806824
// MI455X (gfx1250) — matrix operand order repaired and checked by reading, not compiled, not run on hardware
//
#include <hip/hip_runtime.h>
#include <hip/hip_bf16.h>

typedef _Float16 half_t;
typedef __attribute__((ext_vector_type(16))) _Float16 v16h;
typedef __attribute__((ext_vector_type(8)))  _Float16 v8h;
typedef __attribute__((ext_vector_type(8)))  float    v8f;
typedef __attribute__((ext_vector_type(4)))  unsigned int v4u;
typedef __attribute__((ext_vector_type(8)))  int      v8i;
typedef __attribute__((ext_vector_type(4)))  int      v4i;

#define B__  8
#define C__  256
#define H__  64
#define W__  64
#define L__  (H__ * W__)           // 4096
#define N__  (B__ * L__)           // 32768 tokens
#define DM__ 64
#define DI__ 128
#define DS__ 16
#define EPS__ 1e-5f

#if defined(__has_builtin)
#if __has_builtin(__builtin_amdgcn_tensor_load_to_lds)
#define HAVE_TDM 1
#endif
#endif

__device__ __forceinline__ v16h combine_frag(v8h lo, v8h hi) {
  v16h r;
#pragma unroll
  for (int i = 0; i < 8; ++i) { r[i] = lo[i]; r[i + 8] = hi[i]; }
  return r;
}

enum { EPI_F16 = 0, EPI_F32 = 1, EPI_BN_F16 = 2, EPI_BN_ADD_F32 = 3, EPI_BIAS_TRANS = 4 };

// ---------------------------------------------------------------------------
// WMMA GEMM: block = 8 waves; block stages an NB*16-row weight slab into LDS
// via the Tensor Data Mover, then each wave computes a 16 x (NB*16) strip.
// NB is compile-time (4 normally, 3 for the 48-row x_proj) so all inner loops
// fully unroll with no guards -> no spills.
// Wm is (Krows x K) row-major with ldw == K (contiguous slab per n-group).
// Fragment layouts per CDNA5 ISA 7.12.2 (wave32, f16 A 16x32 / B 32x16).
// ---------------------------------------------------------------------------
template <int EPI, int NB>
__global__ void wmma_gemm_kernel(const half_t* __restrict__ A, int lda,
                                 const half_t* __restrict__ Wm, int K,
                                 int mtiles,
                                 float* __restrict__ outF, half_t* __restrict__ outH, int ldo,
                                 const float* __restrict__ q0, const float* __restrict__ q1,
                                 const float* __restrict__ q2, const float* __restrict__ q3) {
  __shared__ half_t sW[64 * 256];  // up to 64 rows x K<=256, 32 KB

  int mgroups = mtiles >> 3;
  int bm = blockIdx.x % mgroups;       // m-tile group (8 tiles, one per wave)
  int ng = blockIdx.x / mgroups;       // n-group of NB n-tiles
  const int rows = NB * 16;            // weight rows staged per block
  const int elems = rows * K;          // f16 elements (contiguous in Wm)

#ifdef HAVE_TDM
  // ---- TDM: DMA the weight slab (contiguous rows*K f16) into LDS ----------
  if ((threadIdx.x >> 5) == 0) {
    const half_t* gsrc = Wm + (size_t)(ng * rows) * K;
    unsigned long long ga = (unsigned long long)(size_t)gsrc;
    unsigned int lds = (unsigned int)(size_t)(void*)&sW[0];  // low 32 bits = LDS addr
    unsigned int E = (unsigned int)elems;                    // <= 16384
    v4u g0;
    g0[0] = 1u;                                              // count=1, user mode
    g0[1] = lds;                                             // lds_addr
    g0[2] = (unsigned int)(ga & 0xFFFFFFFFu);                // global_addr[31:0]
    g0[3] = (unsigned int)((ga >> 32) & 0x01FFFFFFu) | 0x80000000u;  // addr[56:32] | type=2
    v8i g1;
    g1[0] = 0x00010000;                    // workgroup_mask=0, data_size=1 (2 bytes)
    g1[1] = (int)((E & 0xFFFFu) << 16);    // tensor_dim0 lo16
    g1[2] = (int)((E >> 16) | (1u << 16)); // tensor_dim0 hi | tensor_dim1=1 lo
    g1[3] = (int)((E & 0xFFFFu) << 16);    // tensor_dim1 hi(0) | tile_dim0 = E
    g1[4] = 1;                             // tile_dim1=1, tile_dim2=0
    g1[5] = (int)E;                        // tensor_dim0_stride lo32
    g1[6] = 0;
    g1[7] = 0;
    v4i z4 = {0, 0, 0, 0};
#if __clang_major__ >= 23
    v8i z8 = {0, 0, 0, 0, 0, 0, 0, 0};
    __builtin_amdgcn_tensor_load_to_lds(g0, g1, z4, z4, z8, 0);
#else
    __builtin_amdgcn_tensor_load_to_lds(g0, g1, z4, z4, 0);
#endif
    __builtin_amdgcn_s_wait_tensorcnt(0);
  }
#else
  for (int i = threadIdx.x; i < elems; i += blockDim.x)
    sW[i] = Wm[(size_t)(ng * rows) * K + i];
#endif
  __syncthreads();

  int wv = threadIdx.x >> 5;
  int lane = threadIdx.x & 31;
  int mt = bm * 8 + wv;

  // A: lane<16 -> row m=lane, K {0..7}/{16..23}; lane>=16 -> K {8..15}/{24..31}
  const half_t* aptr = A + (size_t)(mt * 16 + (lane & 15)) * lda + ((lane >> 4) << 3);
  // B from LDS: lane<16 -> col n=lane, K {0..7}/{16..23}; lane>=16 -> K {8..15}/{24..31}
  const half_t* bptr[NB];
#pragma unroll
  for (int nb = 0; nb < NB; ++nb)
    bptr[nb] = sW + (size_t)(nb * 16 + (lane & 15)) * K + ((lane >> 4) << 3);

  v8f acc[NB];
#pragma unroll
  for (int nb = 0; nb < NB; ++nb) acc[nb] = (v8f){};

  int ktiles = K >> 5;
  for (int kt = 0; kt < ktiles; ++kt) {
    v16h a = combine_frag(*(const v8h*)aptr, *(const v8h*)(aptr + 16));
    if (kt + 1 < ktiles) __builtin_prefetch(aptr + 32, 0, 3);  // global_prefetch_b8
    aptr += 32;
#pragma unroll
    for (int nb = 0; nb < NB; ++nb) {
      v16h b = combine_frag(*(const v8h*)bptr[nb], *(const v8h*)(bptr[nb] + 16));
      bptr[nb] += 32;
      acc[nb] = __builtin_amdgcn_wmma_f32_16x16x32_f16(false, a, false, b,
                                                       (short)0, acc[nb], false, false);
    }
  }

  int row0 = mt * 16 + ((lane >> 4) << 3);
#pragma unroll
  for (int nb = 0; nb < NB; ++nb) {
    int col = ng * rows + nb * 16 + (lane & 15);
#pragma unroll
    for (int r = 0; r < 8; ++r) {
      int row = row0 + r;
      float v = acc[nb][r];
      if (EPI == EPI_F16) {
        outH[(size_t)row * ldo + col] = (half_t)v;
      } else if (EPI == EPI_F32) {
        outF[(size_t)row * ldo + col] = v;
      } else if (EPI == EPI_BN_F16 || EPI == EPI_BN_ADD_F32) {
        float sc = rsqrtf(q3[col] + EPS__) * q0[col];
        float w  = (v - q2[col]) * sc + q1[col];
        if (EPI == EPI_BN_F16) outH[(size_t)row * ldo + col] = (half_t)w;
        else                   outF[(size_t)row * ldo + col] += w;
      } else {  // EPI_BIAS_TRANS: store to NCHW float output
        int b = row >> 12;
        int l = row & (L__ - 1);
        outF[((size_t)b * C__ + col) * L__ + l] = v + q0[col];
      }
    }
  }
}

// LayerNorm over C=256; one token per 256-thread block (LDS tree reduction).
template <int FROM_NCHW>
__global__ void layernorm_kernel(const float* __restrict__ src, const float* __restrict__ g,
                                 const float* __restrict__ bt, half_t* __restrict__ dst) {
  int n = blockIdx.x;
  int c = threadIdx.x;
  int b = n >> 12;
  int l = n & (L__ - 1);
  float v = FROM_NCHW ? src[((size_t)b * C__ + c) * L__ + l] : src[(size_t)n * C__ + c];
  __shared__ float s1[C__];
  __shared__ float s2[C__];
  s1[c] = v; s2[c] = v * v;
  __syncthreads();
  for (int off = C__ / 2; off > 0; off >>= 1) {
    if (c < off) { s1[c] += s1[c + off]; s2[c] += s2[c + off]; }
    __syncthreads();
  }
  float mu  = s1[0] * (1.0f / C__);
  float var = s2[0] * (1.0f / C__) - mu * mu;
  float rs  = rsqrtf(var + EPS__);
  dst[(size_t)n * C__ + c] = (half_t)((v - mu) * rs * g[c] + bt[c]);
}

// Causal depthwise conv1d (DC=4) + SiLU on u = xz[:, :128]
__global__ void conv1d_silu_kernel(const half_t* __restrict__ xz, const float* __restrict__ w,
                                   const float* __restrict__ bias, half_t* __restrict__ uo) {
  int idx = blockIdx.x * blockDim.x + threadIdx.x;
  if (idx >= N__ * DI__) return;
  int d = idx & (DI__ - 1);
  int n = idx >> 7;
  int b = n >> 12;
  int l = n & (L__ - 1);
  float acc = bias[d];
#pragma unroll
  for (int k = 0; k < 4; ++k) {
    int ll = l - 3 + k;
    if (ll >= 0) acc += w[d * 4 + k] * (float)xz[(size_t)(b * L__ + ll) * C__ + d];
  }
  acc = acc / (1.0f + __expf(-acc));  // SiLU
  uo[(size_t)n * DI__ + d] = (half_t)acc;
}

// dt = softplus(dbc[:, :4] @ dt_proj_w^T + dt_proj_b)
__global__ void dt_kernel(const float* __restrict__ dbc, const float* __restrict__ dtw,
                          const float* __restrict__ dtbias, float* __restrict__ dt) {
  int idx = blockIdx.x * blockDim.x + threadIdx.x;
  if (idx >= N__ * DI__) return;
  int d = idx & (DI__ - 1);
  int n = idx >> 7;
  float s = dtbias[d];
#pragma unroll
  for (int r = 0; r < 4; ++r) s += dbc[(size_t)n * 48 + r] * dtw[d * 4 + r];
  s = (s > 20.f) ? s : log1pf(__expf(s));
  dt[(size_t)n * DI__ + d] = s;
}

// Selective scan: lane = (d-half, s); 16-lane shfl_xor reduction for y.
__global__ void scan_kernel(const float* __restrict__ dt, const half_t* __restrict__ u,
                            const float* __restrict__ dbc, const float* __restrict__ A_log,
                            float* __restrict__ y) {
  int wave = blockIdx.x * (blockDim.x >> 5) + (threadIdx.x >> 5);  // 0..511
  int lane = threadIdx.x & 31;
  int s  = lane & 15;
  int hf = lane >> 4;
  int b  = wave >> 6;
  int d  = ((wave & 63) << 1) + hf;
  float As = -__expf(A_log[d * DS__ + s]);
  float h  = 0.f;
  const size_t base = (size_t)b * L__;
  for (int l = 0; l < L__; ++l) {
    size_t n  = base + l;
    float dtv = dt[n * DI__ + d];
    float uv  = (float)u[n * DI__ + d];
    float Bt  = dbc[n * 48 + 4 + s];
    float Ct  = dbc[n * 48 + 20 + s];
    h = __expf(dtv * As) * h + dtv * Bt * uv;
    float val = h * Ct;
    val += __shfl_xor(val, 8, 32);
    val += __shfl_xor(val, 4, 32);
    val += __shfl_xor(val, 2, 32);
    val += __shfl_xor(val, 1, 32);
    if (s == 0) y[n * DI__ + d] = val;
  }
}

// y_total = (y + u*D) * silu(z)
__global__ void gate_kernel(const float* __restrict__ y, const half_t* __restrict__ u,
                            const float* __restrict__ Dp, const half_t* __restrict__ xz,
                            half_t* __restrict__ yt) {
  int idx = blockIdx.x * blockDim.x + threadIdx.x;
  if (idx >= N__ * DI__) return;
  int d = idx & (DI__ - 1);
  int n = idx >> 7;
  float yv = y[(size_t)n * DI__ + d] + (float)u[(size_t)n * DI__ + d] * Dp[d];
  float z  = (float)xz[(size_t)n * C__ + DI__ + d];
  yt[(size_t)n * DI__ + d] = (half_t)(yv * (z / (1.f + __expf(-z))));
}

// Dilated 3x3 depthwise conv, zero padding; token-major f16 in/out.
__global__ void dwconv_kernel(const half_t* __restrict__ in, int ldin,
                              const float* __restrict__ w9, half_t* __restrict__ out, int dil) {
  int idx = blockIdx.x * blockDim.x + threadIdx.x;
  if (idx >= N__ * DM__) return;
  int cc = idx & (DM__ - 1);
  int n  = idx >> 6;
  int b  = n >> 12;
  int l  = n & (L__ - 1);
  int h  = l >> 6;
  int x  = l & 63;
  float acc = 0.f;
#pragma unroll
  for (int kh = 0; kh < 3; ++kh) {
    int hh = h + dil * (kh - 1);
    if ((unsigned)hh >= (unsigned)H__) continue;
#pragma unroll
    for (int kw = 0; kw < 3; ++kw) {
      int xx = x + dil * (kw - 1);
      if ((unsigned)xx >= (unsigned)W__) continue;
      acc += w9[cc * 9 + kh * 3 + kw] * (float)in[(size_t)(b * L__ + hh * W__ + xx) * ldin + cc];
    }
  }
  out[(size_t)n * DM__ + cc] = (half_t)acc;
}

__global__ void f32_to_f16_kernel(const float* __restrict__ src, half_t* __restrict__ dst, int n) {
  int i = blockIdx.x * blockDim.x + threadIdx.x;
  if (i < n) dst[i] = (half_t)src[i];
}

// x_proj_w (36,128) -> f16 padded (48,128) so N-tiles are full 16s.
__global__ void xproj_pad_kernel(const float* __restrict__ src, half_t* __restrict__ dst) {
  int i = blockIdx.x * blockDim.x + threadIdx.x;
  if (i >= 48 * 128) return;
  int r = i >> 7;
  dst[i] = (r < 36) ? (half_t)src[i] : (half_t)0.f;
}

extern "C" void kernel_launch(void* const* d_in, const int* in_sizes, int n_in,
                              void* d_out, int out_size, void* d_ws, size_t ws_size,
                              hipStream_t stream) {
  (void)in_sizes; (void)n_in; (void)out_size; (void)ws_size;
  const float* x         = (const float*)d_in[0];
  const float* norm_g    = (const float*)d_in[1];
  const float* norm_bb   = (const float*)d_in[2];
  const float* proj_w    = (const float*)d_in[3];
  const float* proj_b    = (const float*)d_in[4];
  const float* in_proj_w = (const float*)d_in[5];
  const float* conv1d_w  = (const float*)d_in[6];
  const float* conv1d_b  = (const float*)d_in[7];
  const float* x_proj_w  = (const float*)d_in[8];
  const float* dt_proj_w = (const float*)d_in[9];
  const float* dt_proj_b = (const float*)d_in[10];
  const float* A_log     = (const float*)d_in[11];
  const float* Dp        = (const float*)d_in[12];
  const float* out_proj_w= (const float*)d_in[13];
  const float* dw_w      = (const float*)d_in[14];
  const float* pw_w      = (const float*)d_in[15];
  const float* bn_g      = (const float*)d_in[16];
  const float* bn_b      = (const float*)d_in[17];
  const float* bn_mean   = (const float*)d_in[18];
  const float* bn_var    = (const float*)d_in[19];
  float* outp = (float*)d_out;

  // Workspace carve-out (~143 MB total), 256B aligned slabs.
  char* base = (char*)d_ws;
  size_t off = 0;
  auto alloc = [&](size_t bytes) -> void* {
    void* p = base + off;
    off += (bytes + 255) & ~(size_t)255;
    return p;
  };
  half_t* xn_h     = (half_t*)alloc((size_t)N__ * C__ * 2);   // LN1 output (f16)
  half_t* xz_h     = (half_t*)alloc((size_t)N__ * C__ * 2);   // in_proj output u|z
  half_t* u_h      = (half_t*)alloc((size_t)N__ * DI__ * 2);  // conv1d+silu
  float*  dbc      = (float*) alloc((size_t)N__ * 48 * 4);    // x_proj output (padded 48)
  float*  dtb      = (float*) alloc((size_t)N__ * DI__ * 4);  // softplus dt
  float*  ybuf     = (float*) alloc((size_t)N__ * DI__ * 4);  // scan output
  half_t* yt_h     = (half_t*)alloc((size_t)N__ * DI__ * 2);  // gated y
  float*  out1     = (float*) alloc((size_t)N__ * C__ * 4);   // mamba + conv sums
  half_t* dwA      = (half_t*)alloc((size_t)N__ * DM__ * 2);  // conv ping
  half_t* dwB      = (half_t*)alloc((size_t)N__ * DM__ * 2);  // conv pong
  half_t* ln2_h    = (half_t*)alloc((size_t)N__ * C__ * 2);   // LN2 output
  half_t* w_inproj = (half_t*)alloc((size_t)256 * 64 * 2);
  half_t* w_xproj  = (half_t*)alloc((size_t)48 * 128 * 2);
  half_t* w_outprj = (half_t*)alloc((size_t)64 * 128 * 2);
  half_t* w_pw     = (half_t*)alloc((size_t)16 * 64 * 64 * 2);
  half_t* w_proj   = (half_t*)alloc((size_t)256 * 256 * 2);

  // Weight conversions (deterministic, every call).
  f32_to_f16_kernel<<<(256 * 64 + 255) / 256, 256, 0, stream>>>(in_proj_w, w_inproj, 256 * 64);
  xproj_pad_kernel<<<(48 * 128 + 255) / 256, 256, 0, stream>>>(x_proj_w, w_xproj);
  f32_to_f16_kernel<<<(64 * 128 + 255) / 256, 256, 0, stream>>>(out_proj_w, w_outprj, 64 * 128);
  f32_to_f16_kernel<<<(16 * 64 * 64 + 255) / 256, 256, 0, stream>>>(pw_w, w_pw, 16 * 64 * 64);
  f32_to_f16_kernel<<<(256 * 256 + 255) / 256, 256, 0, stream>>>(proj_w, w_proj, 256 * 256);

  // LN1 (NCHW -> token-major f16)
  layernorm_kernel<1><<<N__, 256, 0, stream>>>(x, norm_g, norm_bb, xn_h);

  const int MT = N__ / 16;        // 2048 row tiles
  const int MG = MT / 8;          // 256 m-groups (8 waves per block)
  const int dils[4] = {1, 2, 3, 1};

  for (int c = 0; c < 4; ++c) {
    // in_proj: (N,64) x (256,64)^T -> xz (N,256) f16  [4 n-groups of 64]
    wmma_gemm_kernel<EPI_F16, 4><<<MG * 4, 256, 0, stream>>>(
        xn_h + c * DM__, C__, w_inproj, 64, MT,
        nullptr, xz_h, C__, nullptr, nullptr, nullptr, nullptr);
    // causal conv1d + SiLU on u
    conv1d_silu_kernel<<<(N__ * DI__) / 256, 256, 0, stream>>>(xz_h, conv1d_w, conv1d_b, u_h);
    // x_proj: (N,128) x (48,128)^T -> dbc (N,48) f32  [1 n-group of 48]
    wmma_gemm_kernel<EPI_F32, 3><<<MG, 256, 0, stream>>>(
        u_h, DI__, w_xproj, 128, MT,
        dbc, nullptr, 48, nullptr, nullptr, nullptr, nullptr);
    dt_kernel<<<(N__ * DI__) / 256, 256, 0, stream>>>(dbc, dt_proj_w, dt_proj_b, dtb);
    // sequential selective scan (512 wave32s)
    scan_kernel<<<64, 256, 0, stream>>>(dtb, u_h, dbc, A_log, ybuf);
    gate_kernel<<<(N__ * DI__) / 256, 256, 0, stream>>>(ybuf, u_h, Dp, xz_h, yt_h);
    // out_proj: (N,128) x (64,128)^T -> out1 columns [c*64, c*64+64)
    wmma_gemm_kernel<EPI_F32, 4><<<MG, 256, 0, stream>>>(
        yt_h, DI__, w_outprj, 128, MT,
        out1 + c * DM__, nullptr, C__, nullptr, nullptr, nullptr, nullptr);

    // conv block: 4x (depthwise dilated 3x3 -> pointwise 64x64 WMMA + BN)
    const half_t* cur = xn_h + c * DM__;
    int ldcur = C__;
    for (int s = 0; s < 4; ++s) {
      int si = c * 4 + s;
      dwconv_kernel<<<(N__ * DM__) / 256, 256, 0, stream>>>(
          cur, ldcur, dw_w + (size_t)si * DM__ * 9, dwA, dils[s]);
      if (s < 3) {
        wmma_gemm_kernel<EPI_BN_F16, 4><<<MG, 256, 0, stream>>>(
            dwA, DM__, w_pw + (size_t)si * DM__ * DM__, 64, MT,
            nullptr, dwB, DM__,
            bn_g + si * DM__, bn_b + si * DM__, bn_mean + si * DM__, bn_var + si * DM__);
        cur = dwB; ldcur = DM__;
      } else {  // final stage: BN then accumulate into out1 (mamba + conv)
        wmma_gemm_kernel<EPI_BN_ADD_F32, 4><<<MG, 256, 0, stream>>>(
            dwA, DM__, w_pw + (size_t)si * DM__ * DM__, 64, MT,
            out1 + c * DM__, nullptr, C__,
            bn_g + si * DM__, bn_b + si * DM__, bn_mean + si * DM__, bn_var + si * DM__);
      }
    }
  }

  // LN2 then final 256x256 projection + bias, transposed store to NCHW output.
  layernorm_kernel<0><<<N__, 256, 0, stream>>>(out1, norm_g, norm_bb, ln2_h);
  wmma_gemm_kernel<EPI_BIAS_TRANS, 4><<<MG * 4, 256, 0, stream>>>(
      ln2_h, C__, w_proj, 256, MT,
      outp, nullptr, C__, proj_b, nullptr, nullptr, nullptr);
}
